// GeodesicAttentionBlock_36764920053926
// MI455X (gfx1250) — matrix operand order repaired and checked by reading, not compiled, not run on hardware
//
#include <hip/hip_runtime.h>
#include <hip/hip_bf16.h>
#include <math.h>

#define D_MODEL 256
#define N_HEADS 8
#define D_HEAD  32
#define ROWS    64            // rows of x per workgroup
#define NTHREADS 256
#define NWAVES   8
#define WFRAG_ELEMS 512       // one 32x16 (KxN) bf16 B-fragment: 32 lanes x 16 elems
#define W_ELEMS (8*16*WFRAG_ELEMS)  // 65536 elems per packed 256x256 weight

typedef __bf16 bf16_t;
typedef __attribute__((ext_vector_type(8)))  bf16_t v8bf;
typedef __attribute__((ext_vector_type(16))) bf16_t v16bf;
typedef __attribute__((ext_vector_type(8)))  float  v8f;
typedef __attribute__((ext_vector_type(4)))  unsigned int u32x4;
typedef __attribute__((ext_vector_type(8)))  int i32x8;
typedef __attribute__((ext_vector_type(4)))  int i32x4;

// ---------------------------------------------------------------------------
// Pack a 256x256 fp32 weight (rows = K input dim, cols = N output dim) into
// bf16 WMMA B-fragments. Fragment f = nt*8 + kt holds W[kt*32:+32, nt*16:+16].
// Per-lane element e: lane l -> N = nt*16 + (l&15), K = kt*32 + (l>>4)*8 + (e&7) + (e>>3)*16.
// ---------------------------------------------------------------------------
__global__ void pack_weight_kernel(const float* __restrict__ W,
                                   bf16_t* __restrict__ dst) {
  int idx = blockIdx.x * blockDim.x + threadIdx.x;   // 0 .. 65535
  int f      = idx >> 9;
  int within = idx & 511;
  int lane = within >> 4;
  int e    = within & 15;
  int nt = f >> 3, kt = f & 7;
  int kg = lane >> 4;
  int n = nt * 16 + (lane & 15);
  int k = kt * 32 + kg * 8 + (e & 7) + (e >> 3) * 16;
  dst[idx] = (bf16_t)W[k * D_MODEL + n];
}

// Pack L [8, 32, 32] (per-key-head metric, K-dim = d, N = f) into 16 fragments
// (8 heads x 2 n-tiles), same per-lane B layout as above.
__global__ void pack_L_kernel(const float* __restrict__ L,
                              bf16_t* __restrict__ dst) {
  int idx = blockIdx.x * blockDim.x + threadIdx.x;   // 0 .. 8191
  int f      = idx >> 9;        // 0..15
  int within = idx & 511;
  int lane = within >> 4;
  int e    = within & 15;
  int kh = f >> 1, nt = f & 1;
  int kg = lane >> 4;
  int n = nt * 16 + (lane & 15);
  int d = kg * 8 + (e & 7) + (e >> 3) * 16;
  dst[idx] = (bf16_t)L[kh * D_HEAD * D_HEAD + d * D_HEAD + n];
}

// A-fragment (16x32 bf16, M x K) from a row-major [*,256] bf16 LDS matrix.
// ISA layout: lanes 0-15 row M=l, K = {0..7, 16..23}; lanes 16-31 K offset +8.
__device__ __forceinline__ v16bf ldsA(const bf16_t* m, int rowBase, int colBase,
                                      int lane) {
  int r = lane & 15, kg = lane >> 4;
  const bf16_t* p = m + (rowBase + r) * D_MODEL + colBase + kg * 8;
  v8bf lo = *(const v8bf*)p;
  v8bf hi = *(const v8bf*)(p + 16);
  return __builtin_shufflevector(lo, hi, 0,1,2,3,4,5,6,7,8,9,10,11,12,13,14,15);
}

// A-fragment of (Q - K) for the metric GEMM.
__device__ __forceinline__ v16bf ldsDiff(const bf16_t* q, const bf16_t* k,
                                         int rowBase, int qc, int kc, int lane) {
  int r = lane & 15, kg = lane >> 4;
  const bf16_t* pq = q + (rowBase + r) * D_MODEL + qc + kg * 8;
  const bf16_t* pk = k + (rowBase + r) * D_MODEL + kc + kg * 8;
  v8bf qlo = *(const v8bf*)pq, qhi = *(const v8bf*)(pq + 16);
  v8bf klo = *(const v8bf*)pk, khi = *(const v8bf*)(pk + 16);
  v16bf a;
#pragma unroll
  for (int i = 0; i < 8; ++i) {
    a[i]     = (bf16_t)((float)qlo[i] - (float)klo[i]);
    a[i + 8] = (bf16_t)((float)qhi[i] - (float)khi[i]);
  }
  return a;
}

// Issue a Tensor Data Mover load of an (8 x 256) fp32 tile into LDS.
// D# built per CDNA5 ISA sec. 8.3-8.5 (2-D tensor, data_size = 4 bytes).
// This toolchain's builtin takes 6 args: g0(4), g1(8), g2(4), g3(4), g4(8), cpol.
__device__ __forceinline__ void tdm_load_x(unsigned ldsOff,
                                           unsigned long long ga) {
  ldsOff = (unsigned)__builtin_amdgcn_readfirstlane((int)ldsOff);
  unsigned galo = (unsigned)__builtin_amdgcn_readfirstlane((int)(ga & 0xffffffffull));
  unsigned gahi = (unsigned)__builtin_amdgcn_readfirstlane((int)((ga >> 32) & 0xffffffffull));
  u32x4 g0;
  g0.x = 1u;                                    // count=1, user descriptor
  g0.y = ldsOff;                                // lds_addr (bytes)
  g0.z = galo;                                  // global_addr[31:0]
  g0.w = (gahi & 0x1ffffffu) | (2u << 30);      // global_addr[56:32] | type=2
  i32x8 g1;
  g1[0] = (int)(2u << 16);                      // data_size = 4B, no multicast
  g1[1] = (int)(256u << 16);                    // tensor_dim0 = 256 (lo16 @63:48)
  g1[2] = (int)(0u | (8u << 16));               // dim0 hi | tensor_dim1 = 8
  g1[3] = (int)(0u | (256u << 16));             // dim1 hi | tile_dim0 = 256
  g1[4] = (int)(8u);                            // tile_dim1 = 8, tile_dim2 = 0
  g1[5] = (int)(256u);                          // tensor_dim0_stride = 256
  g1[6] = 0;                                    // stride0 hi | stride1 lo
  g1[7] = 0;
  i32x4 gz4 = {0, 0, 0, 0};
  i32x8 gz8 = {0, 0, 0, 0, 0, 0, 0, 0};
  __builtin_amdgcn_tensor_load_to_lds(g0, g1, gz4, gz4, gz8, 0);
}

__global__ __launch_bounds__(NTHREADS)
void geo_attn_main(const float* __restrict__ x,
                   const bf16_t* __restrict__ pW,   // packed Wq|Wk|Wv|Wout|L
                   const float* __restrict__ b_out,
                   const float* __restrict__ gamma,
                   const float* __restrict__ beta,
                   float* __restrict__ out) {
  extern __shared__ char smem[];
  float*  xs  = (float*) (smem);            //  64x256 f32   (65536 B)
  bf16_t* xbf = (bf16_t*)(smem + 65536);    //  64x256 bf16  (32768 B)
  bf16_t* qbf = (bf16_t*)(smem + 98304);    //  64x256 bf16
  bf16_t* kbf = (bf16_t*)(smem + 131072);   //  64x256 bf16
  float*  vf  = (float*) (smem + 163840);   //  64x256 f32 (reused as h later)
  bf16_t* obf = (bf16_t*)(smem + 229376);   //  64x256 bf16 (attn output)
  float*  sq  = (float*) (smem + 262144);   //  [64][8][8] f32 sq-distances

  const int tid  = threadIdx.x;
  const int lane = tid & 31;
  const int wave = __builtin_amdgcn_readfirstlane(tid >> 5);
  const int blockRow0 = blockIdx.x * ROWS;

  const bf16_t* pWq = pW;
  const bf16_t* pWk = pW + W_ELEMS;
  const bf16_t* pWv = pW + 2 * W_ELEMS;
  const bf16_t* pWo = pW + 3 * W_ELEMS;
  const bf16_t* pL  = pW + 4 * W_ELEMS;

  // ---- Phase 0: TDM async-load this wave's 8-row x slice into LDS ----
  {
    unsigned long long ga =
        (unsigned long long)(x + (size_t)(blockRow0 + wave * 8) * D_MODEL);
    unsigned ldsOff = (unsigned)(unsigned long long)(const void*)
                          (xs + wave * 8 * D_MODEL);
    tdm_load_x(ldsOff, ga);
    __builtin_amdgcn_s_wait_tensorcnt(0);
  }
  __syncthreads();

  // ---- Phase 1: convert x tile to bf16 ----
  for (int i = tid; i < ROWS * D_MODEL; i += NTHREADS)
    xbf[i] = (bf16_t)xs[i];
  __syncthreads();

  // ---- Phase 2: Q/K/V GEMMs (WMMA bf16 -> f32) ----
  for (int j = wave; j < 192; j += NWAVES) {
    int m = j / 64;
    int rem = j - m * 64;
    int mt = rem >> 4, nt = rem & 15;
    const bf16_t* pB = (m == 0 ? pWq : (m == 1 ? pWk : pWv)) +
                       nt * 8 * WFRAG_ELEMS;
    v8f acc = {};
#pragma unroll
    for (int kt = 0; kt < 8; ++kt) {
      v16bf a = ldsA(xbf, mt * 16, kt * 32, lane);
      v16bf b = *(const v16bf*)(pB + kt * WFRAG_ELEMS + lane * 16);
      acc = __builtin_amdgcn_wmma_f32_16x16x32_bf16(false, a, false, b,
                                                    (short)0, acc, false, false);
    }
    int col  = nt * 16 + (lane & 15);
    int rowb = mt * 16 + ((lane >> 4) << 3);
    if (m == 2) {
#pragma unroll
      for (int r = 0; r < 8; ++r) vf[(rowb + r) * D_MODEL + col] = acc[r];
    } else {
      bf16_t* dstm = (m == 0) ? qbf : kbf;
#pragma unroll
      for (int r = 0; r < 8; ++r)
        dstm[(rowb + r) * D_MODEL + col] = (bf16_t)acc[r];
    }
  }
  __syncthreads();

  // ---- Phase 2a: sq_dist[b,q,k] = || L_k^T (Q_q - K_k) ||^2 via WMMA ----
  for (int j = wave; j < 256; j += NWAVES) {
    int kh = j >> 5;
    int rem = j & 31;
    int qh = rem >> 2, mt = rem & 3;
    float part[8];
#pragma unroll
    for (int r = 0; r < 8; ++r) part[r] = 0.f;
    v16bf a = ldsDiff(qbf, kbf, mt * 16, qh * 32, kh * 32, lane);
#pragma unroll
    for (int nt = 0; nt < 2; ++nt) {
      v16bf b = *(const v16bf*)(pL + (kh * 2 + nt) * WFRAG_ELEMS + lane * 16);
      v8f d = {};
      d = __builtin_amdgcn_wmma_f32_16x16x32_bf16(false, a, false, b,
                                                  (short)0, d, false, false);
#pragma unroll
      for (int r = 0; r < 8; ++r) part[r] += d[r] * d[r];
    }
#pragma unroll
    for (int msk = 1; msk <= 8; msk <<= 1) {
#pragma unroll
      for (int r = 0; r < 8; ++r) part[r] += __shfl_xor(part[r], msk, 32);
    }
    if ((lane & 15) == 0) {
      int rowb = mt * 16 + ((lane >> 4) << 3);
#pragma unroll
      for (int r = 0; r < 8; ++r)
        sq[(rowb + r) * 64 + qh * 8 + kh] = part[r];
    }
  }
  __syncthreads();

  // ---- Phase 2b: softmax over 8 keys + attn @ V (fp32 VALU) ----
  {
    int row = tid >> 2;
    int q0  = tid & 3;
    const float inv = 0.17677669529663687f;  // 1/sqrt(32)
    for (int qq = 0; qq < 2; ++qq) {
      int qh = q0 + qq * 4;
      float l[8], mx = -1e30f;
#pragma unroll
      for (int k = 0; k < 8; ++k) {
        l[k] = -sq[row * 64 + qh * 8 + k] * inv;
        mx = fmaxf(mx, l[k]);
      }
      float w[8], s = 0.f;
#pragma unroll
      for (int k = 0; k < 8; ++k) { w[k] = __expf(l[k] - mx); s += w[k]; }
      float rs = 1.f / s;
#pragma unroll
      for (int k = 0; k < 8; ++k) w[k] *= rs;
      for (int d = 0; d < D_HEAD; ++d) {
        float o = 0.f;
#pragma unroll
        for (int k = 0; k < 8; ++k)
          o += w[k] * vf[row * D_MODEL + k * D_HEAD + d];
        obf[row * D_MODEL + qh * D_HEAD + d] = (bf16_t)o;
      }
    }
  }
  __syncthreads();

  // ---- Phase 3: out @ Wout + b_out + residual (WMMA) ----
  float* hf = vf;  // V no longer needed; reuse as h
  for (int j = wave; j < 64; j += NWAVES) {
    int mt = j >> 4, nt = j & 15;
    v8f acc = {};
#pragma unroll
    for (int kt = 0; kt < 8; ++kt) {
      v16bf a = ldsA(obf, mt * 16, kt * 32, lane);
      v16bf b = *(const v16bf*)(pWo + (nt * 8 + kt) * WFRAG_ELEMS + lane * 16);
      acc = __builtin_amdgcn_wmma_f32_16x16x32_bf16(false, a, false, b,
                                                    (short)0, acc, false, false);
    }
    int col  = nt * 16 + (lane & 15);
    int rowb = mt * 16 + ((lane >> 4) << 3);
    float bo = b_out[col];
#pragma unroll
    for (int r = 0; r < 8; ++r) {
      int row = rowb + r;
      hf[row * D_MODEL + col] = acc[r] + bo + xs[row * D_MODEL + col];
    }
  }
  __syncthreads();

  // ---- Phase 4: LayerNorm (4 lanes per row, shuffle-reduce) ----
  {
    int row = tid >> 2;
    int seg = tid & 3;
    const float* hr = hf + row * D_MODEL;
    float s = 0.f, ss = 0.f;
    for (int c = seg * 64; c < seg * 64 + 64; ++c) {
      float v = hr[c];
      s += v;
      ss += v * v;
    }
    s  += __shfl_xor(s, 1, 32);  ss += __shfl_xor(ss, 1, 32);
    s  += __shfl_xor(s, 2, 32);  ss += __shfl_xor(ss, 2, 32);
    float mean = s * (1.f / 256.f);
    float var  = ss * (1.f / 256.f) - mean * mean;
    float rstd = rsqrtf(var + 1e-5f);
    float* orow = out + (size_t)(blockRow0 + row) * D_MODEL;
    for (int c = seg * 64; c < seg * 64 + 64; ++c)
      orow[c] = (hr[c] - mean) * rstd * gamma[c] + beta[c];
  }
}

extern "C" void kernel_launch(void* const* d_in, const int* in_sizes, int n_in,
                              void* d_out, int out_size, void* d_ws,
                              size_t ws_size, hipStream_t stream) {
  (void)in_sizes; (void)n_in; (void)out_size; (void)ws_size;
  const float* x    = (const float*)d_in[0];
  const float* Wq   = (const float*)d_in[1];
  const float* Wk   = (const float*)d_in[2];
  const float* Wv   = (const float*)d_in[3];
  const float* Wout = (const float*)d_in[4];
  const float* b_o  = (const float*)d_in[5];
  const float* L    = (const float*)d_in[6];
  const float* gam  = (const float*)d_in[7];
  const float* bet  = (const float*)d_in[8];
  bf16_t* pW = (bf16_t*)d_ws;   // needs 270336 bf16 = ~528 KB of workspace

  pack_weight_kernel<<<256, 256, 0, stream>>>(Wq,   pW);
  pack_weight_kernel<<<256, 256, 0, stream>>>(Wk,   pW + W_ELEMS);
  pack_weight_kernel<<<256, 256, 0, stream>>>(Wv,   pW + 2 * W_ELEMS);
  pack_weight_kernel<<<256, 256, 0, stream>>>(Wout, pW + 3 * W_ELEMS);
  pack_L_kernel<<<32, 256, 0, stream>>>(L, pW + 4 * W_ELEMS);

  size_t smem = 278528;  // 272 KB of the 320 KB/WGP
  geo_attn_main<<<65536 / ROWS, NTHREADS, smem, stream>>>(
      x, pW, b_o, gam, bet, (float*)d_out);
}
